// Temporal_Inter_SA_25031069401493
// MI455X (gfx1250) — compile-verified, matrix operand order checked by reading, not run on hardware
//
#include <hip/hip_runtime.h>
#include <hip/hip_bf16.h>
#include <math.h>

// ---------------------------------------------------------------------------
// Types
// ---------------------------------------------------------------------------
typedef __bf16 bf16_t;
typedef bf16_t v16bf __attribute__((ext_vector_type(16)));
typedef float  v8f   __attribute__((ext_vector_type(8)));
typedef unsigned int uint4v __attribute__((ext_vector_type(4)));

union FragBF {
    uint4v u[2];
    v16bf  b;
};

// Problem constants
#define NTOK   40960        // B*F*H*W = 2*5*64*64
#define CCH    256
#define HWSZ   4096         // H*W
#define FDIM   5

// ---------------------------------------------------------------------------
// Weight packing: fp32 -> bf16, kept in [O][K] (= [N][K]) row-major layout,
// which is exactly the fragment-friendly layout the GEMM wants for B.
// ---------------------------------------------------------------------------
__global__ __launch_bounds__(256) void pack_whv_kernel(
    const float* __restrict__ wh, const float* __restrict__ wv,
    const float* __restrict__ bh, const float* __restrict__ bv,
    bf16_t* __restrict__ whv, float* __restrict__ bqkv)
{
    int idx = blockIdx.x * 256 + threadIdx.x;
    if (idx < 768) bqkv[idx] = (idx < 384) ? bh[idx] : bv[idx - 384];
    if (idx >= 768 * 256) return;
    int o = idx / 256;       // output channel (N)
    int k = idx % 256;       // input channel  (K)
    float v;
    if (o < 384) v = (k < 128)  ? wh[o * 128 + k] : 0.0f;                 // horizontal block
    else         v = (k >= 128) ? wv[(o - 384) * 128 + (k - 128)] : 0.0f; // vertical block
    whv[idx] = (bf16_t)v;
}

__global__ __launch_bounds__(256) void pack_cvt_kernel(
    const float* __restrict__ w, bf16_t* __restrict__ dst, int n)
{
    int idx = blockIdx.x * 256 + threadIdx.x;
    if (idx < n) dst[idx] = (bf16_t)w[idx];
}

// ---------------------------------------------------------------------------
// LN1: x (b,f,c,h,w) strided -> token-major bf16 [NTOK][256]
// ---------------------------------------------------------------------------
__global__ __launch_bounds__(256) void ln1_kernel(
    const float* __restrict__ x, const float* __restrict__ g,
    const float* __restrict__ bta, bf16_t* __restrict__ xn)
{
    int token = blockIdx.x * 256 + threadIdx.x;
    int bfi = token >> 12;
    int hw  = token & 4095;
    const float* base = x + (size_t)bfi * CCH * HWSZ + hw;   // + c*4096
    float s = 0.f, ss = 0.f;
    for (int c = 0; c < CCH; ++c) {
        float v = base[(size_t)c * HWSZ];
        s += v; ss += v * v;
    }
    float m   = s * (1.0f / CCH);
    float var = ss * (1.0f / CCH) - m * m;
    float r   = rsqrtf(var + 1e-5f);
    bf16_t* o = xn + (size_t)token * CCH;
    for (int c = 0; c < CCH; ++c) {
        float v = (base[(size_t)c * HWSZ] - m) * r * g[c] + bta[c];
        o[c] = (bf16_t)v;
    }
}

// ---------------------------------------------------------------------------
// LN2: token-major fp32 [NTOK][256] -> token-major bf16. One wave per token.
// ---------------------------------------------------------------------------
__global__ __launch_bounds__(256) void ln2_kernel(
    const float* __restrict__ y, const float* __restrict__ g,
    const float* __restrict__ bta, bf16_t* __restrict__ tn)
{
    int token = blockIdx.x * 8 + (threadIdx.x >> 5);
    int lane  = threadIdx.x & 31;
    const float* p = y + (size_t)token * CCH + lane * 8;
    float v[8];
#pragma unroll
    for (int i = 0; i < 8; ++i) v[i] = p[i];
    float s = 0.f, ss = 0.f;
#pragma unroll
    for (int i = 0; i < 8; ++i) { s += v[i]; ss += v[i] * v[i]; }
#pragma unroll
    for (int off = 16; off; off >>= 1) {
        s  += __shfl_xor(s,  off, 32);
        ss += __shfl_xor(ss, off, 32);
    }
    float m   = s * (1.0f / CCH);
    float var = ss * (1.0f / CCH) - m * m;
    float r   = rsqrtf(var + 1e-5f);
    bf16_t* o = tn + (size_t)token * CCH + lane * 8;
#pragma unroll
    for (int i = 0; i < 8; ++i) {
        int c = lane * 8 + i;
        o[i] = (bf16_t)((v[i] - m) * r * g[c] + bta[c]);
    }
}

// ---------------------------------------------------------------------------
// Attention: temporal (seq len F=5), d = 1024 per group.
// 2048 groups: [0,1024) horizontal (b,n,h), [1024,2048) vertical (b,n,w).
// ---------------------------------------------------------------------------
__global__ __launch_bounds__(256) void attn_kernel(
    const bf16_t* __restrict__ qkv, bf16_t* __restrict__ ao)
{
    __shared__ float sred[8 * 25];
    __shared__ float sp[25];

    int g    = blockIdx.x;
    bool vert = g >= 1024;
    int gg   = vert ? g - 1024 : g;
    int b    = gg >> 9;
    int n    = (gg >> 6) & 7;
    int l    = gg & 63;                      // h (horiz) or w (vert)
    int tid  = threadIdx.x;
    int wave = tid >> 5, lane = tid & 31;
    int voff  = (vert ? 384 : 0) + n * 16;   // qkv row channel base (q)
    int ochan = (vert ? 128 : 0) + n * 16;

    float s[5][5];
#pragma unroll
    for (int i = 0; i < 5; ++i)
#pragma unroll
        for (int j = 0; j < 5; ++j) s[i][j] = 0.f;

    // pass 1: scores
    for (int dc = tid; dc < 1024; dc += 256) {
        int pos = dc >> 4;
        int e   = dc & 15;
        int h   = vert ? pos : l;
        int w   = vert ? l : pos;
        float qv[5], kv[5];
#pragma unroll
        for (int f = 0; f < FDIM; ++f) {
            size_t tok = (((size_t)b * FDIM + f) * 64 + h) * 64 + w;
            const bf16_t* p = qkv + tok * 768 + voff + e;
            qv[f] = (float)p[0];
            kv[f] = (float)p[128];
        }
#pragma unroll
        for (int qi = 0; qi < 5; ++qi)
#pragma unroll
            for (int ki = 0; ki < 5; ++ki) s[qi][ki] += qv[qi] * kv[ki];
    }

    // reduce 25 partials across block
#pragma unroll
    for (int i = 0; i < 25; ++i) {
        float v = s[i / 5][i % 5];
#pragma unroll
        for (int off = 16; off; off >>= 1) v += __shfl_xor(v, off, 32);
        if (lane == 0) sred[wave * 25 + i] = v;
    }
    __syncthreads();
    if (tid < 25) {
        float tot = 0.f;
#pragma unroll
        for (int wv = 0; wv < 8; ++wv) tot += sred[wv * 25 + tid];
        sred[tid] = tot * 0.03125f;          // 1/sqrt(1024)
    }
    __syncthreads();
    if (tid < 5) {
        float mx = -1e30f;
#pragma unroll
        for (int k = 0; k < 5; ++k) mx = fmaxf(mx, sred[tid * 5 + k]);
        float e[5], sum = 0.f;
#pragma unroll
        for (int k = 0; k < 5; ++k) { e[k] = __expf(sred[tid * 5 + k] - mx); sum += e[k]; }
        float inv = 1.f / sum;
#pragma unroll
        for (int k = 0; k < 5; ++k) sp[tid * 5 + k] = e[k] * inv;
    }
    __syncthreads();

    float pm[5][5];
#pragma unroll
    for (int qi = 0; qi < 5; ++qi)
#pragma unroll
        for (int ki = 0; ki < 5; ++ki) pm[qi][ki] = sp[qi * 5 + ki];

    // pass 2: P*V
    for (int dc = tid; dc < 1024; dc += 256) {
        int pos = dc >> 4;
        int e   = dc & 15;
        int h   = vert ? pos : l;
        int w   = vert ? l : pos;
        float vv[5];
#pragma unroll
        for (int f = 0; f < FDIM; ++f) {
            size_t tok = (((size_t)b * FDIM + f) * 64 + h) * 64 + w;
            vv[f] = (float)qkv[tok * 768 + voff + 256 + e];
        }
#pragma unroll
        for (int f = 0; f < FDIM; ++f) {
            float o = 0.f;
#pragma unroll
            for (int k = 0; k < 5; ++k) o += pm[f][k] * vv[k];
            size_t tok = (((size_t)b * FDIM + f) * 64 + h) * 64 + w;
            ao[tok * 256 + ochan + e] = (bf16_t)o;
        }
    }
}

// ---------------------------------------------------------------------------
// Tiled bf16 WMMA GEMM with async global->LDS double-buffered pipeline.
//   out = A[M][K] * B[N][K]^T + bias (+ epilogue)
// BM=BN=128, BK=32, 256 threads = 8 waves, each wave computes 64x32.
// Both A and B tiles are staged with GLOBAL_LOAD_ASYNC_TO_LDS_B128 (ASYNCcnt),
// overlapping the next tile's fill with the current tile's WMMAs.
// EPI 0: bf16 store (bias)
// EPI 1: bf16 store (bias + exact GELU)
// EPI 2: fp32 store token-major (bias + residual gathered from x strided layout)
// EPI 3: fp32 scatter to (b,f,c,h,w) (bias + add token-major fp32 residual)
// ---------------------------------------------------------------------------
#define LDSB     40                       // padded row stride in halfs (80 B)
#define TILE_A_B (128 * LDSB * 2)         // 10240 B per matrix tile
#define BUF_B    (2 * TILE_A_B)           // 20480 B per buffer (A then B)
#define SMEM_B   (2 * BUF_B)              // 40960 B total (double buffered)

template <int EPI>
__global__ __launch_bounds__(256) void gemm_bf16_kernel(
    const bf16_t* __restrict__ A, const bf16_t* __restrict__ Bm,
    const float* __restrict__ bias, void* __restrict__ out,
    const float* __restrict__ extra, int M, int N, int K)
{
    extern __shared__ __align__(16) char smem[];   // dynamic LDS: offset 0

    const int tid  = threadIdx.x;
    const int wave = tid >> 5;
    const int lane = tid & 31;
    const int m0 = blockIdx.x * 128;
    const int n0 = blockIdx.y * 128;
    const int wm = (wave & 1) * 64;       // wave row offset in tile
    const int wn = (wave >> 1) * 32;      // wave col offset in tile
    const int lhalf = lane >> 4;          // 0/1 -> K halves
    const int l16   = lane & 15;

    v8f acc[4][2];
#pragma unroll
    for (int i = 0; i < 4; ++i)
#pragma unroll
        for (int j = 0; j < 2; ++j)
#pragma unroll
            for (int r = 0; r < 8; ++r) acc[i][j][r] = 0.f;

    // issue async copies of one (A,B) K-tile into LDS buffer `buf`
    auto issue_tile = [&](int kk, int buf) {
        unsigned base = (unsigned)(buf * BUF_B);
#pragma unroll
        for (int it = 0; it < 2; ++it) {
            int task = tid + it * 256;
            int row = task >> 2, ch = task & 3;
            unsigned la = base + row * (LDSB * 2) + ch * 16;
            unsigned long long ga = (unsigned long long)(const char*)
                (A + (size_t)(m0 + row) * K + kk + ch * 8);
            asm volatile("global_load_async_to_lds_b128 %0, %1, off"
                         :: "v"(la), "v"(ga) : "memory");
            unsigned lb = base + (unsigned)TILE_A_B + row * (LDSB * 2) + ch * 16;
            unsigned long long gb = (unsigned long long)(const char*)
                (Bm + (size_t)(n0 + row) * K + kk + ch * 8);
            asm volatile("global_load_async_to_lds_b128 %0, %1, off"
                         :: "v"(lb), "v"(gb) : "memory");
        }
    };

    issue_tile(0, 0);
    int buf = 0;
    for (int kk = 0; kk < K; kk += 32) {
        // current buffer's async fills done (this wave) ...
        asm volatile("s_wait_asynccnt 0x0" ::: "memory");
        // ... and all waves done reading the buffer we are about to refill
        __syncthreads();
        if (kk + 32 < K) issue_tile(kk + 32, buf ^ 1);

        const bf16_t* sA = (const bf16_t*)(smem + buf * BUF_B);
        const bf16_t* sB = (const bf16_t*)(smem + buf * BUF_B + TILE_A_B);

        FragBF fa[4], fb[2];
#pragma unroll
        for (int i = 0; i < 4; ++i) {
            const bf16_t* p = &sA[(wm + i * 16 + l16) * LDSB + lhalf * 16];
            fa[i].u[0] = *(const uint4v*)p;
            fa[i].u[1] = *(const uint4v*)(p + 8);
        }
#pragma unroll
        for (int j = 0; j < 2; ++j) {
            const bf16_t* p = &sB[(wn + j * 16 + l16) * LDSB + lhalf * 16];
            fb[j].u[0] = *(const uint4v*)p;
            fb[j].u[1] = *(const uint4v*)(p + 8);
        }
#pragma unroll
        for (int i = 0; i < 4; ++i)
#pragma unroll
            for (int j = 0; j < 2; ++j)
                acc[i][j] = __builtin_amdgcn_wmma_f32_16x16x32_bf16(
                    false, fa[i].b, false, fb[j].b,
                    (short)0, acc[i][j], false, false);
        buf ^= 1;
    }

    // epilogue: VGPR r holds row (r + 8*lhalf) of the 16x16 tile, col = l16
#pragma unroll
    for (int i = 0; i < 4; ++i)
#pragma unroll
        for (int j = 0; j < 2; ++j) {
            int col = n0 + wn + j * 16 + l16;
            float bcol = bias[col];
#pragma unroll
            for (int r = 0; r < 8; ++r) {
                int row = m0 + wm + i * 16 + lhalf * 8 + r;
                float v = acc[i][j][r] + bcol;
                if (EPI == 0) {
                    ((bf16_t*)out)[(size_t)row * N + col] = (bf16_t)v;
                } else if (EPI == 1) {
                    v = 0.5f * v * (1.0f + erff(v * 0.70710678118654752f));
                    ((bf16_t*)out)[(size_t)row * N + col] = (bf16_t)v;
                } else if (EPI == 2) {
                    int bfi = row >> 12, hw = row & 4095;
                    v += extra[((size_t)bfi * CCH + col) * HWSZ + hw];
                    ((float*)out)[(size_t)row * N + col] = v;
                } else {
                    v += extra[(size_t)row * CCH + col];
                    int bfi = row >> 12, hw = row & 4095;
                    ((float*)out)[((size_t)bfi * CCH + col) * HWSZ + hw] = v;
                }
            }
        }
}

// ---------------------------------------------------------------------------
// Launch.  Workspace layout (needs ~170 MB):
//   [0,        80 MB)  qkv bf16 [N][768]  -> later reused as MLP hidden [N][1024]
//   [80 MB,   100 MB)  xn  bf16 [N][256]  -> later reused as LN2 out
//   [100 MB,  120 MB)  ao  bf16 [N][256]
//   [120 MB,  160 MB)  y   fp32 [N][256]
//   [160 MB, +1.6 MB)  packed weights ([N][K] bf16) + qkv bias
// ---------------------------------------------------------------------------
extern "C" void kernel_launch(void* const* d_in, const int* in_sizes, int n_in,
                              void* d_out, int out_size, void* d_ws, size_t ws_size,
                              hipStream_t stream)
{
    const float* x     = (const float*)d_in[0];
    const float* ln1_g = (const float*)d_in[1];
    const float* ln1_b = (const float*)d_in[2];
    const float* wh    = (const float*)d_in[3];
    const float* bh    = (const float*)d_in[4];
    const float* wv    = (const float*)d_in[5];
    const float* bv    = (const float*)d_in[6];
    const float* wf    = (const float*)d_in[7];
    const float* bf_   = (const float*)d_in[8];
    const float* ln2_g = (const float*)d_in[9];
    const float* ln2_b = (const float*)d_in[10];
    const float* w1    = (const float*)d_in[11];
    const float* b1    = (const float*)d_in[12];
    const float* w2    = (const float*)d_in[13];
    const float* b2    = (const float*)d_in[14];

    char* ws = (char*)d_ws;
    bf16_t* qkv_hid = (bf16_t*)(ws);                            // 83,886,080 B
    bf16_t* xn_tn   = (bf16_t*)(ws + 83886080ULL);              // 20,971,520 B
    bf16_t* ao      = (bf16_t*)(ws + 104857600ULL);             // 20,971,520 B
    float*  ytok    = (float*) (ws + 125829120ULL);             // 41,943,040 B
    char*   wb      =          ws + 167772160ULL;
    bf16_t* whv  = (bf16_t*)(wb);                               // 393,216 B [768][256]
    bf16_t* wfB  = (bf16_t*)(wb + 393216);                      // 131,072 B [256][256]
    bf16_t* w1B  = (bf16_t*)(wb + 524288);                      // 524,288 B [1024][256]
    bf16_t* w2B  = (bf16_t*)(wb + 1048576);                     // 524,288 B [256][1024]
    float*  bqkv = (float*) (wb + 1572864);                     //   3,072 B

    // 1) pack weights (bf16, [N][K] row-major)
    pack_whv_kernel<<<(768 * 256 + 255) / 256, 256, 0, stream>>>(wh, wv, bh, bv, whv, bqkv);
    pack_cvt_kernel<<<(256 * 256 + 255) / 256, 256, 0, stream>>>(wf, wfB, 256 * 256);
    pack_cvt_kernel<<<(1024 * 256 + 255) / 256, 256, 0, stream>>>(w1, w1B, 1024 * 256);
    pack_cvt_kernel<<<(256 * 1024 + 255) / 256, 256, 0, stream>>>(w2, w2B, 256 * 1024);

    // 2) LN1 -> xn bf16
    ln1_kernel<<<NTOK / 256, 256, 0, stream>>>(x, ln1_g, ln1_b, xn_tn);

    // 3) QKV GEMM (block-diagonal wh/wv): [N,256]x[256,768]
    gemm_bf16_kernel<0><<<dim3(NTOK / 128, 768 / 128), 256, SMEM_B, stream>>>(
        xn_tn, whv, bqkv, qkv_hid, nullptr, NTOK, 768, 256);

    // 4) temporal attention (horizontal + vertical)
    attn_kernel<<<2048, 256, 0, stream>>>(qkv_hid, ao);

    // 5) output proj + residual(x) -> y fp32 token-major
    gemm_bf16_kernel<2><<<dim3(NTOK / 128, 256 / 128), 256, SMEM_B, stream>>>(
        ao, wfB, bf_, ytok, x, NTOK, 256, 256);

    // 6) LN2 -> tn bf16 (reuse xn region)
    ln2_kernel<<<NTOK / 8, 256, 0, stream>>>(ytok, ln2_g, ln2_b, xn_tn);

    // 7) MLP fc1 + GELU -> hidden bf16 (reuse qkv region)
    gemm_bf16_kernel<1><<<dim3(NTOK / 128, 1024 / 128), 256, SMEM_B, stream>>>(
        xn_tn, w1B, b1, qkv_hid, nullptr, NTOK, 1024, 256);

    // 8) MLP fc2 + residual(y) -> scatter fp32 to (b,f,c,h,w)
    gemm_bf16_kernel<3><<<dim3(NTOK / 128, 256 / 128), 256, SMEM_B, stream>>>(
        qkv_hid, w2B, b2, d_out, ytok, NTOK, 256, 1024);
}
